// S5Dual_89653147337062
// MI455X (gfx1250) — matrix operand order repaired and checked by reading, not compiled, not run on hardware
//
#include <hip/hip_runtime.h>
#include <hip/hip_bf16.h>
#include <math.h>

#define DEVI __device__ __forceinline__

typedef __attribute__((ext_vector_type(16))) __bf16 v16bf;
typedef __attribute__((ext_vector_type(8)))  __bf16 v8bf;
typedef __attribute__((ext_vector_type(8)))  float  v8f;
typedef __attribute__((ext_vector_type(4)))  int    v4i;

#if __has_builtin(__builtin_amdgcn_global_load_async_to_lds_b128) && \
    __has_builtin(__builtin_amdgcn_s_wait_asynccnt)
#define USE_ASYNC_LDS 1
typedef __attribute__((address_space(1))) v4i gv4i;   // global int4
typedef __attribute__((address_space(3))) v4i sv4i;   // LDS int4
#else
#define USE_ASYNC_LDS 0
#endif

constexpr int Bn = 4, Ln = 8192, Hn = 256, Pn = 256;
constexpr int LC = 32;                 // chunk length (rows per workgroup)
constexpr int NC = Ln / LC;            // 256 chunks per sequence
constexpr int XROW = Hn + 8;           // padded LDS row stride (bf16 elems)
constexpr int UROW = Pn + 8;

// ---- LDS carving (one block; f32 stage aliases bur/bui) ----
constexpr int SM_XTILE = 0;                        // LC*XROW bf16  = 16896 B
constexpr int SM_BUR   = SM_XTILE + LC * XROW * 2; // LC*UROW bf16  = 16896 B
constexpr int SM_BUI   = SM_BUR   + LC * UROW * 2; // LC*UROW bf16  = 16896 B
constexpr int SM_STAGE = SM_BUR;                   // LC*Hn f32     = 32768 B (aliases bur+bui)
constexpr int SM_BYTES = SM_BUI + LC * UROW * 2;   // 50688 B

// ---- workspace layout (bytes) ----
constexpr size_t WS_LAM   = 0;                                  // [2][Pn] float2
constexpr size_t WS_LAMLC = WS_LAM   + (size_t)2*Pn*8;          // [2][Pn] float2
constexpr size_t WS_PB    = WS_LAMLC + (size_t)2*Pn*8;          // [2][2][Hn*Pn] bf16 packed
constexpr size_t WS_PC    = WS_PB    + (size_t)2*2*Hn*Pn*2;     // [2][2][Pn*Hn] bf16 packed
constexpr size_t WS_S     = WS_PC    + (size_t)2*2*Pn*Hn*2;     // [2][Bn][NC][Pn] float2
constexpr size_t WS_PRE   = WS_S     + (size_t)2*Bn*NC*Pn*8;    // same size

struct DirParams {
  const float *log_real, *imag, *log_Delta, *B_real, *B_imag, *C_real, *C_imag, *D;
};
struct AllParams {
  DirParams d[2];
  const float* x;
  float* out;
  char* ws;
};

// Pack (k,n) element of a K x N bf16 WMMA B-operand so that each lane's
// fragment is 16 contiguous bf16 values (two b128 loads).
// B layout used here (32x16, contiguous k, A matches): lane = half*16 + n%16, element e -> K = e + half*16.
DEVI int packIndex(int k, int n) {
  int kk = k >> 5, r = k & 31, half = r >> 4, e = r & 15;
  int nn = n >> 4, lo = n & 15;
  int lane = half * 16 + lo;
  return (((kk * 16 + nn) * 32 + lane) << 4) + e;
}

union AFrag { v16bf v; v8bf h[2]; };
union BF4 { __bf16 b[4]; uint2 u; };

// ---------------- K0a: discretize + pack B_bar ----------------
__global__ __launch_bounds__(256) void k_discretize_packB(AllParams P) {
  int dir = blockIdx.x;
  int p = threadIdx.x;
  const DirParams& dp = P.d[dir];
  float Lr = -expf(dp.log_real[p]);
  float Li = dp.imag[p];
  float Delta = expf(dp.log_Delta[p]);
  float ad = Lr * Delta, bd = Li * Delta;
  float ea = expf(ad);
  float lbr = ea * cosf(bd), lbi = ea * sinf(bd);
  float denom = fmaxf(Lr * Lr + Li * Li, 1e-12f);
  float invr = Lr / denom, invi = -Li / denom;
  float dr = lbr - 1.0f, di = lbi;
  float cr = invr * dr - invi * di;
  float ci = invr * di + invi * dr;

  ((float2*)(P.ws + WS_LAM))[dir * Pn + p] = make_float2(lbr, lbi);
  float eL = expf((float)LC * ad);
  ((float2*)(P.ws + WS_LAMLC))[dir * Pn + p] =
      make_float2(eL * cosf((float)LC * bd), eL * sinf((float)LC * bd));

  __bf16* pbr = (__bf16*)(P.ws + WS_PB) + (size_t)(dir * 2 + 0) * Hn * Pn;
  __bf16* pbi = (__bf16*)(P.ws + WS_PB) + (size_t)(dir * 2 + 1) * Hn * Pn;
  for (int h = 0; h < Hn; ++h) {
    float br = dp.B_real[p * Hn + h], bi = dp.B_imag[p * Hn + h];
    float bbr = cr * br - ci * bi;
    float bbi = cr * bi + ci * br;
    int idx = packIndex(h, p);      // GEMM1 B-operand: K=h, N=p
    pbr[idx] = (__bf16)bbr;
    pbi[idx] = (__bf16)bbi;
  }
}

// ---------------- K0b: pack C (imag negated) ----------------
__global__ __launch_bounds__(256) void k_packC(AllParams P) {
  int t = blockIdx.x * 256 + threadIdx.x;   // 2*Hn*Pn
  int dir = t / (Hn * Pn);
  int rem = t % (Hn * Pn);
  int h = rem / Pn, p = rem % Pn;
  const DirParams& dp = P.d[dir];
  __bf16* pcr = (__bf16*)(P.ws + WS_PC) + (size_t)(dir * 2 + 0) * Pn * Hn;
  __bf16* pci = (__bf16*)(P.ws + WS_PC) + (size_t)(dir * 2 + 1) * Pn * Hn;
  int idx = packIndex(p, h);                // GEMM2 B-operand: K=p, N=h
  pcr[idx] = (__bf16)dp.C_real[h * Pn + p];
  pci[idx] = (__bf16)(-dp.C_imag[h * Pn + p]);
}

// ---------------- shared helpers ----------------
// Stage the fp32 x chunk into LDS (async DMA path when available), then
// convert to bf16 into xtile with packed 8-byte LDS stores.
DEVI void load_x_tile(const float* x, int b, int c, int dir,
                      __bf16* xtile, float* stage) {
  int t = threadIdx.x;
#if USE_ASYNC_LDS
  #pragma unroll
  for (int j = 0; j < 8; ++j) {
    int idx = t + 256 * j;          // 2048 float4's
    int row = idx >> 6;
    int col4 = idx & 63;
    int lf = c * LC + row;
    int g = dir ? (Ln - 1 - lf) : lf;
    const float* gp = x + ((size_t)b * Ln + g) * Hn + col4 * 4;
    float* lp = stage + (size_t)idx * 4;
    __builtin_amdgcn_global_load_async_to_lds_b128((gv4i*)gp, (sv4i*)lp, 0, 0);
  }
  __builtin_amdgcn_s_wait_asynccnt(0);
  __syncthreads();
  #pragma unroll
  for (int j = 0; j < 8; ++j) {
    int idx = t + 256 * j;
    int row = idx >> 6;
    int col4 = idx & 63;
    const float4 vv = ((const float4*)stage)[idx];
    BF4 q;
    q.b[0] = (__bf16)vv.x; q.b[1] = (__bf16)vv.y;
    q.b[2] = (__bf16)vv.z; q.b[3] = (__bf16)vv.w;
    *(uint2*)(xtile + row * XROW + col4 * 4) = q.u;
  }
#else
  #pragma unroll
  for (int j = 0; j < 8; ++j) {
    int idx = t + 256 * j;
    int row = idx >> 6;
    int col4 = idx & 63;
    int lf = c * LC + row;
    int g = dir ? (Ln - 1 - lf) : lf;
    const float4 vv = ((const float4*)(x + ((size_t)b * Ln + g) * Hn))[col4];
    BF4 q;
    q.b[0] = (__bf16)vv.x; q.b[1] = (__bf16)vv.y;
    q.b[2] = (__bf16)vv.z; q.b[3] = (__bf16)vv.w;
    *(uint2*)(xtile + row * XROW + col4 * 4) = q.u;
  }
#endif
}

// GEMM1: Bu[l,p] = sum_h x[l,h] * Bbar[p,h]  (both real and imag RHS)
DEVI void gemm1(const __bf16* xtile, const __bf16* pbr, const __bf16* pbi,
                __bf16* bur, __bf16* bui) {
  int w = threadIdx.x >> 5, lane = threadIdx.x & 31;
  int rt = w & 1, ngrp = w >> 1;
  int lane16 = lane & 15, half = lane >> 4;
  int m = rt * 16 + lane16;

  AFrag a[8];
  #pragma unroll
  for (int ks = 0; ks < 8; ++ks) {
    int kb = ks * 32 + half * 16;
    a[ks].h[0] = *(const v8bf*)(xtile + m * XROW + kb);
    a[ks].h[1] = *(const v8bf*)(xtile + m * XROW + kb + 8);
  }
  #pragma unroll
  for (int nl = 0; nl < 4; ++nl) {
    int nn = ngrp * 4 + nl;
    v8f accR = {};
    v8f accI = {};
    #pragma unroll
    for (int ks = 0; ks < 8; ++ks) {
      size_t boff = (size_t)((ks * 16 + nn) * 32 + lane) * 16;
      AFrag br, bi;
      br.h[0] = *(const v8bf*)(pbr + boff);
      br.h[1] = *(const v8bf*)(pbr + boff + 8);
      bi.h[0] = *(const v8bf*)(pbi + boff);
      bi.h[1] = *(const v8bf*)(pbi + boff + 8);
      accR = __builtin_amdgcn_wmma_f32_16x16x32_bf16(false, a[ks].v, false, br.v,
                                                     (short)0, accR, false, false);
      accI = __builtin_amdgcn_wmma_f32_16x16x32_bf16(false, a[ks].v, false, bi.v,
                                                     (short)0, accI, false, false);
    }
    #pragma unroll
    for (int r = 0; r < 8; ++r) {
      int row = rt * 16 + half * 8 + r;
      int col = nn * 16 + lane16;
      bur[row * UROW + col] = (__bf16)accR[r];
      bui[row * UROW + col] = (__bf16)accI[r];
    }
  }
}

// ---------------- K1: per-chunk final state ----------------
__global__ __launch_bounds__(256) void k_chunk_state(AllParams P) {
  __shared__ alignas(16) char smem[SM_BYTES];
  __bf16* xtile = (__bf16*)(smem + SM_XTILE);
  __bf16* bur   = (__bf16*)(smem + SM_BUR);
  __bf16* bui   = (__bf16*)(smem + SM_BUI);
  float*  stage = (float*)(smem + SM_STAGE);

  int bid = blockIdx.x;
  int dir = bid / (Bn * NC);
  int rem = bid % (Bn * NC);
  int b = rem / NC, c = rem % NC;

  const __bf16* pb = (const __bf16*)(P.ws + WS_PB) + (size_t)dir * 2 * Hn * Pn;
  __builtin_prefetch(pb + (threadIdx.x << 6), 0, 1);

  load_x_tile(P.x, b, c, dir, xtile, stage);
  __syncthreads();
  gemm1(xtile, pb, pb + (size_t)Hn * Pn, bur, bui);
  __syncthreads();

  int p = threadIdx.x;
  float2 lam = ((const float2*)(P.ws + WS_LAM))[dir * Pn + p];
  float sr = 0.f, si = 0.f;
  for (int l = 0; l < LC; ++l) {
    float ur = (float)bur[l * UROW + p];
    float ui = (float)bui[l * UROW + p];
    float nr = lam.x * sr - lam.y * si + ur;
    float ni = lam.x * si + lam.y * sr + ui;
    sr = nr; si = ni;
  }
  ((float2*)(P.ws + WS_S))[((size_t)(dir * Bn + b) * NC + c) * Pn + p] =
      make_float2(sr, si);
}

// ---------------- K2: inter-chunk exclusive prefix ----------------
__global__ __launch_bounds__(256) void k_chunk_prefix(AllParams P) {
  int dir = blockIdx.x / Bn, b = blockIdx.x % Bn;
  int p = threadIdx.x;
  float2 lam = ((const float2*)(P.ws + WS_LAMLC))[dir * Pn + p];
  const float2* S = (const float2*)(P.ws + WS_S) + (size_t)(dir * Bn + b) * NC * Pn;
  float2* pre = (float2*)(P.ws + WS_PRE) + (size_t)(dir * Bn + b) * NC * Pn;
  float pr = 0.f, pi = 0.f;
  for (int c = 0; c < NC; ++c) {
    pre[c * Pn + p] = make_float2(pr, pi);
    float2 s = S[c * Pn + p];
    float nr = lam.x * pr - lam.y * pi + s.x;
    float ni = lam.x * pi + lam.y * pr + s.y;
    pr = nr; pi = ni;
  }
}

// ---------------- K3: recompute + seeded scan + GEMM2 + epilogue ----------------
__global__ __launch_bounds__(256) void k_output(AllParams P, int dir) {
  __shared__ alignas(16) char smem[SM_BYTES];
  __bf16* xtile = (__bf16*)(smem + SM_XTILE);
  __bf16* bur   = (__bf16*)(smem + SM_BUR);
  __bf16* bui   = (__bf16*)(smem + SM_BUI);
  float*  stage = (float*)(smem + SM_STAGE);

  int bid = blockIdx.x;
  int b = bid / NC, c = bid % NC;

  const __bf16* pb  = (const __bf16*)(P.ws + WS_PB) + (size_t)dir * 2 * Hn * Pn;
  const __bf16* pcr = (const __bf16*)(P.ws + WS_PC) + (size_t)(dir * 2 + 0) * Pn * Hn;
  const __bf16* pci = pcr + (size_t)Pn * Hn;
  __builtin_prefetch(pb  + (threadIdx.x << 6), 0, 1);
  __builtin_prefetch(pcr + (threadIdx.x << 6), 0, 1);

  load_x_tile(P.x, b, c, dir, xtile, stage);
  __syncthreads();
  gemm1(xtile, pb, pb + (size_t)Hn * Pn, bur, bui);
  __syncthreads();

  {
    int p = threadIdx.x;
    float2 lam = ((const float2*)(P.ws + WS_LAM))[dir * Pn + p];
    float2 s0 = ((const float2*)(P.ws + WS_PRE))
                    [((size_t)(dir * Bn + b) * NC + c) * Pn + p];
    float sr = s0.x, si = s0.y;
    for (int l = 0; l < LC; ++l) {
      float ur = (float)bur[l * UROW + p];
      float ui = (float)bui[l * UROW + p];
      float nr = lam.x * sr - lam.y * si + ur;
      float ni = lam.x * si + lam.y * sr + ui;
      sr = nr; si = ni;
      bur[l * UROW + p] = (__bf16)sr;   // xs_r in place
      bui[l * UROW + p] = (__bf16)si;   // xs_i in place
    }
  }
  __syncthreads();

  int w = threadIdx.x >> 5, lane = threadIdx.x & 31;
  int rt = w & 1, ngrp = w >> 1;
  int lane16 = lane & 15, half = lane >> 4;
  int m = rt * 16 + lane16;
  const float* Dv = P.d[dir].D;

  #pragma unroll
  for (int nl = 0; nl < 4; ++nl) {
    int nn = ngrp * 4 + nl;
    v8f acc = {};
    #pragma unroll
    for (int ks = 0; ks < 8; ++ks) {
      int kb = ks * 32 + half * 16;
      AFrag ar, ai;
      ar.h[0] = *(const v8bf*)(bur + m * UROW + kb);
      ar.h[1] = *(const v8bf*)(bur + m * UROW + kb + 8);
      ai.h[0] = *(const v8bf*)(bui + m * UROW + kb);
      ai.h[1] = *(const v8bf*)(bui + m * UROW + kb + 8);
      size_t boff = (size_t)((ks * 16 + nn) * 32 + lane) * 16;
      AFrag cr, ci;
      cr.h[0] = *(const v8bf*)(pcr + boff);
      cr.h[1] = *(const v8bf*)(pcr + boff + 8);
      ci.h[0] = *(const v8bf*)(pci + boff);
      ci.h[1] = *(const v8bf*)(pci + boff + 8);
      acc = __builtin_amdgcn_wmma_f32_16x16x32_bf16(false, ar.v, false, cr.v,
                                                    (short)0, acc, false, false);
      acc = __builtin_amdgcn_wmma_f32_16x16x32_bf16(false, ai.v, false, ci.v,
                                                    (short)0, acc, false, false);
    }
    int hcol = nn * 16 + lane16;
    float dcoef = Dv[hcol];
    #pragma unroll
    for (int r = 0; r < 8; ++r) {
      int lf = c * LC + rt * 16 + half * 8 + r;
      int g = dir ? (Ln - 1 - lf) : lf;
      size_t addr = ((size_t)b * Ln + g) * Hn + hcol;
      float xv = P.x[addr];
      float v = acc[r] + dcoef * xv;
      v = 0.5f * v * (1.0f + erff(v * 0.70710678118654752f));
      if (dir == 0) P.out[addr] = v;
      else          P.out[addr] += v;     // fwd launch completed first -> no race
    }
  }
}

extern "C" void kernel_launch(void* const* d_in, const int* in_sizes, int n_in,
                              void* d_out, int out_size, void* d_ws, size_t ws_size,
                              hipStream_t stream) {
  (void)in_sizes; (void)n_in; (void)out_size; (void)ws_size;
  AllParams P;
  P.x = (const float*)d_in[0];
  for (int dir = 0; dir < 2; ++dir) {
    int base = 1 + dir * 8;
    P.d[dir].log_real  = (const float*)d_in[base + 0];
    P.d[dir].imag      = (const float*)d_in[base + 1];
    P.d[dir].log_Delta = (const float*)d_in[base + 2];
    P.d[dir].B_real    = (const float*)d_in[base + 3];
    P.d[dir].B_imag    = (const float*)d_in[base + 4];
    P.d[dir].C_real    = (const float*)d_in[base + 5];
    P.d[dir].C_imag    = (const float*)d_in[base + 6];
    P.d[dir].D         = (const float*)d_in[base + 7];
  }
  P.out = (float*)d_out;
  P.ws  = (char*)d_ws;

  k_discretize_packB<<<2, 256, 0, stream>>>(P);
  k_packC<<<(2 * Hn * Pn) / 256, 256, 0, stream>>>(P);
  k_chunk_state<<<2 * Bn * NC, 256, 0, stream>>>(P);
  k_chunk_prefix<<<2 * Bn, 256, 0, stream>>>(P);
  k_output<<<Bn * NC, 256, 0, stream>>>(P, 0);
  k_output<<<Bn * NC, 256, 0, stream>>>(P, 1);
}
